// SNNMiddleLayer_54941221651200
// MI455X (gfx1250) — matrix operand order repaired and checked by reading, not compiled, not run on hardware
//
#include <hip/hip_runtime.h>

// ---------------------------------------------------------------------------
// SNNMiddleLayer for MI455X (gfx1250): 4 chained GEMMs in bf16 WMMA with fp32
// accumulation; final GEMM fuses the residual + scale epilogue.
//   X = HS (8192x2048) @ w_down^T (2048x512)
//   H = X @ w1^T,  O = H @ w2^T   (512x512)
//   out = resid + scale * (O @ w_up^T (512x2048))
// Wave tile 32x64 (2x4 v_wmma_f32_16x16x32_bf16), block tile 128x128,
// 8 waves / 256 threads per block. 12 global_load_b128 feed 8 WMMAs per
// 32-K step (~21 lane-FLOP per fetched byte); all operands are L2-resident.
// ---------------------------------------------------------------------------

typedef __attribute__((ext_vector_type(16))) __bf16 v16bf;
typedef __attribute__((ext_vector_type(8)))  float  v8f;

union Frag {
  v16bf v;
  uint4 q[2];
};

__device__ __forceinline__ unsigned bf_round(float f) {
  // round-to-nearest-even fp32 -> bf16 (as u16 in low bits)
  unsigned u = __builtin_bit_cast(unsigned, f);
  return (u + 0x7FFFu + ((u >> 16) & 1u)) >> 16;
}

// --------------------------- fp32 -> bf16 convert ---------------------------
__global__ __launch_bounds__(256) void cvt_f32_bf16(const float* __restrict__ in,
                                                    unsigned short* __restrict__ out,
                                                    long n) {
  long i = ((long)blockIdx.x * blockDim.x + threadIdx.x) * 4;
  if (i >= n) return;
  float4 f = *reinterpret_cast<const float4*>(in + i);
  uint2 p;
  p.x = bf_round(f.x) | (bf_round(f.y) << 16);
  p.y = bf_round(f.z) | (bf_round(f.w) << 16);
  *reinterpret_cast<uint2*>(out + i) = p;
}

// ------------------------------- WMMA GEMM ----------------------------------
// C[M x N] = A[M x K] * W^T, with W stored [N x K] (nn.Linear layout). Both A
// and W rows are K-contiguous bf16, mapping directly onto the CDNA5 16-bit
// A/B fragment VGPR layouts (b128 loads only, 16-byte aligned).
template <bool FINAL>
__global__ __launch_bounds__(256) void gemm_bf16_wmma(
    const unsigned short* __restrict__ A,
    const unsigned short* __restrict__ W,
    unsigned short* __restrict__ Obf,      // used when !FINAL
    const float* __restrict__ resid,       // used when FINAL
    const float* __restrict__ scale_p,     // used when FINAL
    float* __restrict__ Of,                // used when FINAL
    int K, int N) {
  const int tid  = threadIdx.x;
  const int wave = tid >> 5;
  const int lane = tid & 31;
  const int wm   = wave & 3;   // 4 waves down M  (32 rows each)
  const int wn   = wave >> 2;  // 2 waves across N (64 cols each)
  const int m0   = blockIdx.x * 128 + wm * 32;
  const int n0   = blockIdx.y * 128 + wn * 64;
  const int r    = lane & 15;  // row owner within 16
  const int h    = lane >> 4;  // half-wave selector per ISA fragment layout

  // A fragment: lane (r,h) holds A[m][k + 8h + 0..7] and A[m][k+16+8h + 0..7]
  const unsigned short* a0p = A + (size_t)(m0 + r) * K + h * 8;
  const unsigned short* a1p = a0p + (size_t)16 * K;
  // B fragment: lane (r,h) holds W[n][k + 8h + 0..7] and W[n][k+16+8h + 0..7]
  const unsigned short* bp  = W + (size_t)(n0 + r) * K + h * 8;
  const size_t bstep = (size_t)16 * K;  // 16 N-rows per fragment

  v8f c[2][4];
#pragma unroll
  for (int tm = 0; tm < 2; ++tm)
#pragma unroll
    for (int tn = 0; tn < 4; ++tn) c[tm][tn] = (v8f){};

  for (int k = 0; k < K; k += 32) {
    __builtin_prefetch((const void*)(a0p + k + 256), 0, 3);
    __builtin_prefetch((const void*)(bp  + k + 256), 0, 3);

    Frag a0, a1, b[4];
    a0.q[0] = *reinterpret_cast<const uint4*>(a0p + k);
    a0.q[1] = *reinterpret_cast<const uint4*>(a0p + k + 16);
    a1.q[0] = *reinterpret_cast<const uint4*>(a1p + k);
    a1.q[1] = *reinterpret_cast<const uint4*>(a1p + k + 16);
#pragma unroll
    for (int tn = 0; tn < 4; ++tn) {
      const unsigned short* bt = bp + bstep * tn + k;
      b[tn].q[0] = *reinterpret_cast<const uint4*>(bt);
      b[tn].q[1] = *reinterpret_cast<const uint4*>(bt + 16);
    }

#pragma unroll
    for (int tn = 0; tn < 4; ++tn) {
      c[0][tn] = __builtin_amdgcn_wmma_f32_16x16x32_bf16(
          false, a0.v, false, b[tn].v, (short)0, c[0][tn], false, false);
      c[1][tn] = __builtin_amdgcn_wmma_f32_16x16x32_bf16(
          false, a1.v, false, b[tn].v, (short)0, c[1][tn], false, false);
    }
  }

  // Epilogue. C fragment layout: VGPR j, lanes 0-15 -> (M=j,  N=lane),
  //                              lanes 16-31        -> (M=j+8,N=lane-16).
  const int n_lane = lane & 15;
  const float scale = FINAL ? *scale_p : 0.0f;
#pragma unroll
  for (int tm = 0; tm < 2; ++tm) {
#pragma unroll
    for (int tn = 0; tn < 4; ++tn) {
#pragma unroll
      for (int j = 0; j < 8; ++j) {
        const int m = m0 + tm * 16 + 8 * h + j;
        const int n = n0 + tn * 16 + n_lane;
        const size_t idx = (size_t)m * N + n;
        const float val = c[tm][tn][j];
        if (FINAL) {
          Of[idx] = resid[idx] + scale * val;
        } else {
          Obf[idx] = (unsigned short)bf_round(val);
        }
      }
    }
  }
}

// ------------------------------- launcher -----------------------------------
extern "C" void kernel_launch(void* const* d_in, const int* in_sizes, int n_in,
                              void* d_out, int out_size, void* d_ws, size_t ws_size,
                              hipStream_t stream) {
  (void)in_sizes; (void)n_in; (void)out_size; (void)ws_size;

  const float* hs      = (const float*)d_in[0];  // [4,2048,2048]
  const float* w_down  = (const float*)d_in[1];  // [512,2048]
  const float* w_up    = (const float*)d_in[2];  // [2048,512]
  const float* w1      = (const float*)d_in[3];  // [512,512]
  const float* w2      = (const float*)d_in[4];  // [512,512]
  const float* scale_p = (const float*)d_in[5];  // scalar
  float* out = (float*)d_out;

  const long M  = 4L * 2048;  // 8192 tokens
  const long QH = 2048;
  const long SH = 512;

  // Workspace layout (bf16 as u16). Total ~56 MB.
  unsigned short* ws    = (unsigned short*)d_ws;
  unsigned short* hs_bf = ws;                  // M * QH
  unsigned short* wd_bf = hs_bf + M * QH;      // SH * QH
  unsigned short* wu_bf = wd_bf + SH * QH;     // QH * SH
  unsigned short* w1_bf = wu_bf + QH * SH;     // SH * SH
  unsigned short* w2_bf = w1_bf + SH * SH;     // SH * SH
  unsigned short* buf0  = w2_bf + SH * SH;     // M * SH (X, then O)
  unsigned short* buf1  = buf0 + M * SH;       // M * SH (H)

  auto cvt = [&](const float* src, unsigned short* dst, long n) {
    long blocks = (n / 4 + 255) / 256;
    cvt_f32_bf16<<<dim3((unsigned)blocks), dim3(256), 0, stream>>>(src, dst, n);
  };
  cvt(hs,     hs_bf, M * QH);
  cvt(w_down, wd_bf, SH * QH);
  cvt(w_up,   wu_bf, QH * SH);
  cvt(w1,     w1_bf, SH * SH);
  cvt(w2,     w2_bf, SH * SH);

  const dim3 blk(256);
  const unsigned gx = (unsigned)(M / 128);  // 64

  // X = HS @ w_down^T          (K=2048, N=512)
  gemm_bf16_wmma<false><<<dim3(gx, (unsigned)(SH / 128)), blk, 0, stream>>>(
      hs_bf, wd_bf, buf0, nullptr, nullptr, nullptr, (int)QH, (int)SH);
  // H = X @ w1^T               (K=512, N=512)
  gemm_bf16_wmma<false><<<dim3(gx, (unsigned)(SH / 128)), blk, 0, stream>>>(
      buf0, w1_bf, buf1, nullptr, nullptr, nullptr, (int)SH, (int)SH);
  // O = H @ w2^T               (K=512, N=512)
  gemm_bf16_wmma<false><<<dim3(gx, (unsigned)(SH / 128)), blk, 0, stream>>>(
      buf1, w2_bf, buf0, nullptr, nullptr, nullptr, (int)SH, (int)SH);
  // out = resid + scale * (O @ w_up^T)   (K=512, N=2048)
  gemm_bf16_wmma<true><<<dim3(gx, (unsigned)(QH / 128)), blk, 0, stream>>>(
      buf0, wu_bf, nullptr, hs, scale_p, out, (int)SH, (int)QH);
}
